// GCN_OH_1614907703640
// MI455X (gfx1250) — compile-verified, matrix operand order checked by reading, not run on hardware
//
#include <hip/hip_runtime.h>

#define F_DIM 512

typedef __bf16        bf16_t;
typedef bf16_t        v16bf __attribute__((ext_vector_type(16)));
typedef float         v8f   __attribute__((ext_vector_type(8)));
typedef float         v4f   __attribute__((ext_vector_type(4)));
typedef unsigned int  v4u   __attribute__((ext_vector_type(4)));

union FragU { v16bf v; v4u u[2]; };

__device__ __forceinline__ unsigned short f2bf(float f) {
    unsigned u = __float_as_uint(f);
    u += 0x7FFFu + ((u >> 16) & 1u);          // round-to-nearest-even
    return (unsigned short)(u >> 16);
}

// ---------------- init ----------------
__global__ void k_zero4(float* __restrict__ p, int total4) {
    int t = blockIdx.x * blockDim.x + threadIdx.x;
    if (t >= total4) return;
    v4f z = {0.f, 0.f, 0.f, 0.f};
    *(v4f*)(p + (size_t)t * 4) = z;
}

__global__ void k_init_small(int* __restrict__ degi, float* __restrict__ agg2, int n) {
    int t = blockIdx.x * blockDim.x + threadIdx.x;
    if (t >= n) return;
    degi[t] = 1;          // self-loop contributes 1 to every degree
    agg2[t] = 0.f;
}

// ---------------- bf16 conversion ----------------
__global__ void k_cvt_x(const float* __restrict__ x, unsigned short* __restrict__ xb, int total4) {
    int t = blockIdx.x * blockDim.x + threadIdx.x;
    if (t >= total4) return;
    v4f v = *(const v4f*)(x + (size_t)t * 4);
    unsigned p0 = (unsigned)f2bf(v.x) | ((unsigned)f2bf(v.y) << 16);
    unsigned p1 = (unsigned)f2bf(v.z) | ((unsigned)f2bf(v.w) << 16);
    unsigned* d = (unsigned*)(xb + (size_t)t * 4);
    d[0] = p0; d[1] = p1;
}

// W1 is [K=512][N=512] row-major; produce bf16 transposed w1t[N][K] so B-fragment
// loads are contiguous along K per lane.
__global__ void k_cvt_w1t(const float* __restrict__ W1, unsigned short* __restrict__ w1t) {
    int t = blockIdx.x * blockDim.x + threadIdx.x;
    if (t >= 512 * 512) return;
    int nrow = t >> 9;
    int k    = t & 511;
    w1t[(size_t)nrow * 512 + k] = f2bf(W1[(size_t)k * 512 + nrow]);
}

// ---------------- degree / norm ----------------
__global__ void k_deg(const int* __restrict__ dstv, int* __restrict__ degi, int E) {
    int t = blockIdx.x * blockDim.x + threadIdx.x;
    if (t >= E) return;
    atomicAdd(&degi[dstv[t]], 1);
}

__global__ void k_dinv(const int* __restrict__ degi, float* __restrict__ dinv, int n) {
    int t = blockIdx.x * blockDim.x + threadIdx.x;
    if (t >= n) return;
    float d = (float)degi[t];
    dinv[t] = (d > 0.f) ? rsqrtf(d) : 0.f;
}

// ---------------- GEMM1: h1 = x @ W1 via bf16 WMMA, f32 accumulate ----------------
// One 16x16 output tile per wave32, K stepped by 32 (16 WMMA ops per tile).
// Fragment layout per CDNA5 ISA 16-bit A-matrix table:
//   lanes 0-15 : elements 0-7 -> K = k0+0..7,  elements 8-15 -> K = k0+16..23
//   lanes 16-31: elements 0-7 -> K = k0+8..15, elements 8-15 -> K = k0+24..31
// -> per lane, two contiguous 16B loads at (k0 + 8*half) and (+16).
__global__ void __launch_bounds__(256) k_gemm1(const unsigned short* __restrict__ xb,
                                               const unsigned short* __restrict__ w1t,
                                               float* __restrict__ h1, int nTilesM) {
    int lane = threadIdx.x & 31;
    int wv   = threadIdx.x >> 5;
    int wgid = blockIdx.x * 8 + wv;
    int tileN = wgid & 31;      // 512/16 = 32 column tiles
    int tileM = wgid >> 5;
    if (tileM >= nTilesM) return;   // wave-uniform: EXEC stays all-ones for WMMA

    int mr  = (tileM << 4) + (lane & 15);
    int nc  = (tileN << 4) + (lane & 15);
    int hlf = lane >> 4;

    const unsigned short* aP = xb  + (size_t)mr * 512 + 8 * hlf;
    const unsigned short* bP = w1t + (size_t)nc * 512 + 8 * hlf;

    v8f acc = {0.f, 0.f, 0.f, 0.f, 0.f, 0.f, 0.f, 0.f};
    for (int k0 = 0; k0 < 512; k0 += 32) {
        FragU a, b;
        a.u[0] = *(const v4u*)(aP + k0);
        a.u[1] = *(const v4u*)(aP + k0 + 16);
        b.u[0] = *(const v4u*)(bP + k0);
        b.u[1] = *(const v4u*)(bP + k0 + 16);
        __builtin_prefetch(aP + k0 + 64, 0, 1);   // global_prefetch_b8
        __builtin_prefetch(bP + k0 + 64, 0, 1);
        acc = __builtin_amdgcn_wmma_f32_16x16x32_bf16(
                  false, a.v, false, b.v, (short)0, acc, false, false);
    }

    // C/D layout: VGPR r -> M = r (lanes 0-15) / r+8 (lanes 16-31), N = lane&15
    int mbase = (tileM << 4) + (hlf << 3);
    #pragma unroll
    for (int r = 0; r < 8; ++r)
        h1[(size_t)(mbase + r) * 512 + nc] = acc[r];
}

// ---------------- aggregation layer 1: agg[dst] += dinv[s]*dinv[d] * h1[src] ----------------
__global__ void k_agg1(const int* __restrict__ srcv, const int* __restrict__ dstv,
                       const float* __restrict__ dinv, const float* __restrict__ h1,
                       float* __restrict__ agg, int E) {
    long long gid = (long long)blockIdx.x * blockDim.x + threadIdx.x;
    int e = (int)(gid >> 7);                  // 128 threads per edge (512 feats / 4)
    if (e >= E) return;
    int f = ((int)gid & 127) << 2;
    int s = srcv[e], d = dstv[e];
    float nrm = dinv[s] * dinv[d];
    v4f v = *(const v4f*)(h1 + (size_t)s * 512 + f);
    float* base = agg + (size_t)d * 512 + f;
    unsafeAtomicAdd(base + 0, nrm * v.x);     // hardware global_atomic_add_f32
    unsafeAtomicAdd(base + 1, nrm * v.y);
    unsafeAtomicAdd(base + 2, nrm * v.z);
    unsafeAtomicAdd(base + 3, nrm * v.w);
}

// self-loop + bias + ReLU fused, in-place into agg (becomes layer-1 output)
__global__ void k_fin1(const float* __restrict__ h1, const float* __restrict__ dinv,
                       const float* __restrict__ b1, float* __restrict__ agg, int total4) {
    int t = blockIdx.x * blockDim.x + threadIdx.x;
    if (t >= total4) return;
    size_t idx = (size_t)t * 4;
    int node = (int)(idx >> 9);
    int f    = (int)(idx & 511);
    float di = dinv[node];
    float sl = di * di;                        // self-loop norm
    v4f a  = *(const v4f*)(agg + idx);
    v4f h  = *(const v4f*)(h1 + idx);
    v4f bb = *(const v4f*)(b1 + f);
    v4f r;
    r.x = fmaxf(a.x + sl * h.x + bb.x, 0.f);
    r.y = fmaxf(a.y + sl * h.y + bb.y, 0.f);
    r.z = fmaxf(a.z + sl * h.z + bb.z, 0.f);
    r.w = fmaxf(a.w + sl * h.w + bb.w, 0.f);
    *(v4f*)(agg + idx) = r;
}

// ---------------- layer 2 matvec: h2[i] = dot(out1[i,:], W2[:,0]) ----------------
__global__ void k_gemm2(const float* __restrict__ out1, const float* __restrict__ W2,
                        float* __restrict__ h2, int n) {
    int lane = threadIdx.x & 31;
    int wave = (blockIdx.x * blockDim.x + threadIdx.x) >> 5;
    if (wave >= n) return;
    const float* row = out1 + (size_t)wave * 512;
    float s = 0.f;
    for (int f = lane * 4; f < 512; f += 128) {
        v4f v = *(const v4f*)(row + f);
        v4f w = *(const v4f*)(W2 + f);
        s += v.x * w.x + v.y * w.y + v.z * w.z + v.w * w.w;
    }
    #pragma unroll
    for (int m = 16; m >= 1; m >>= 1) s += __shfl_xor(s, m, 32);
    if (lane == 0) h2[wave] = s;
}

__global__ void k_agg2(const int* __restrict__ srcv, const int* __restrict__ dstv,
                       const float* __restrict__ dinv, const float* __restrict__ h2,
                       float* __restrict__ agg2, int E) {
    int t = blockIdx.x * blockDim.x + threadIdx.x;
    if (t >= E) return;
    int s = srcv[t], d = dstv[t];
    unsafeAtomicAdd(&agg2[d], dinv[s] * dinv[d] * h2[s]);
}

// finalize layer 2 (+b2, mean over axis of size 1 == identity) and ohe = o @ Wl + bl
__global__ void k_fin2(const float* __restrict__ agg2, const float* __restrict__ dinv,
                       const float* __restrict__ h2, const float* __restrict__ b2,
                       const float* __restrict__ o, const float* __restrict__ Wl,
                       const float* __restrict__ bl,
                       float* __restrict__ hfin, float* __restrict__ ohe, int n) {
    int t = blockIdx.x * blockDim.x + threadIdx.x;
    if (t >= n) return;
    float di = dinv[t];
    hfin[t] = agg2[t] + di * di * h2[t] + b2[0];
    v4f ov = *(const v4f*)(o + (size_t)t * 4);
    ohe[t] = ov.x * Wl[0] + ov.y * Wl[1] + ov.z * Wl[2] + ov.w * Wl[3] + bl[0];
}

// ---------------- broadcast-add: out[i][j] = hfin[j] + ohe[i]  (256 MB stream) ----------------
__global__ void k_bcast(const float* __restrict__ hfin, const float* __restrict__ ohe,
                        float* __restrict__ out, int n) {
    long long gid = (long long)blockIdx.x * blockDim.x + threadIdx.x;
    int row4 = n >> 2;
    long long total = (long long)n * row4;
    if (gid >= total) return;
    int i  = (int)(gid / row4);
    int j4 = (int)(gid % row4) << 2;
    float oi = ohe[i];
    v4f h = *(const v4f*)(hfin + j4);
    v4f r = {h.x + oi, h.y + oi, h.z + oi, h.w + oi};
    __builtin_nontemporal_store(r, (v4f*)(out + (size_t)i * n + j4));  // bypass-cache stream
}

// ---------------- host launcher ----------------
extern "C" void kernel_launch(void* const* d_in, const int* in_sizes, int n_in,
                              void* d_out, int out_size, void* d_ws, size_t ws_size,
                              hipStream_t stream) {
    (void)n_in; (void)out_size; (void)ws_size;
    const float* x  = (const float*)d_in[0];
    const float* o  = (const float*)d_in[1];
    const int*   ei = (const int*)d_in[2];
    const float* W1 = (const float*)d_in[3];
    const float* b1 = (const float*)d_in[4];
    const float* W2 = (const float*)d_in[5];
    const float* b2 = (const float*)d_in[6];
    const float* Wl = (const float*)d_in[7];
    const float* bl = (const float*)d_in[8];
    float* out = (float*)d_out;

    const int n = in_sizes[0] / F_DIM;    // 8192
    const int E = in_sizes[2] / 2;        // 262144
    const int* srcv = ei;
    const int* dstv = ei + E;

    // carve workspace (~43 MB), 256B-aligned regions
    char* ws = (char*)d_ws;
    size_t off = 0;
    auto carve = [&](size_t bytes) -> char* {
        char* p = ws + off;
        off = (off + bytes + 255) & ~(size_t)255;
        return p;
    };
    unsigned short* xb   = (unsigned short*)carve((size_t)n * 512 * 2);
    unsigned short* w1t  = (unsigned short*)carve((size_t)512 * 512 * 2);
    float* h1    = (float*)carve((size_t)n * 512 * 4);
    float* agg1  = (float*)carve((size_t)n * 512 * 4);   // reused as layer-1 output
    int*   degi  = (int*)carve((size_t)n * 4);
    float* dinv  = (float*)carve((size_t)n * 4);
    float* h2    = (float*)carve((size_t)n * 4);
    float* agg2  = (float*)carve((size_t)n * 4);
    float* hfin  = (float*)carve((size_t)n * 4);
    float* ohe   = (float*)carve((size_t)n * 4);

    const int B = 256;
    int total4 = n * 512 / 4;

    k_zero4     <<<(total4 + B - 1) / B, B, 0, stream>>>(agg1, total4);
    k_init_small<<<(n + B - 1) / B,      B, 0, stream>>>(degi, agg2, n);
    k_cvt_x     <<<(total4 + B - 1) / B, B, 0, stream>>>(x, xb, total4);
    k_cvt_w1t   <<<(512 * 512 + B - 1) / B, B, 0, stream>>>(W1, w1t);
    k_deg       <<<(E + B - 1) / B,      B, 0, stream>>>(dstv, degi, E);
    k_dinv      <<<(n + B - 1) / B,      B, 0, stream>>>(degi, dinv, n);

    int tilesM = n / 16;
    int waves  = tilesM * 32;
    k_gemm1<<<(waves + 7) / 8, B, 0, stream>>>(xb, w1t, h1, tilesM);

    long long aggThreads = (long long)E * 128;
    k_agg1<<<(unsigned)((aggThreads + B - 1) / B), B, 0, stream>>>(srcv, dstv, dinv, h1, agg1, E);
    k_fin1<<<(total4 + B - 1) / B, B, 0, stream>>>(h1, dinv, b1, agg1, total4);

    k_gemm2<<<(n * 32 + B - 1) / B, B, 0, stream>>>(agg1, W2, h2, n);
    k_agg2 <<<(E + B - 1) / B,      B, 0, stream>>>(srcv, dstv, dinv, h2, agg2, E);
    k_fin2 <<<(n + B - 1) / B,      B, 0, stream>>>(agg2, dinv, h2, b2, o, Wl, bl, hfin, ohe, n);

    long long bThreads = (long long)n * (n / 4);
    k_bcast<<<(unsigned)((bThreads + B - 1) / B), B, 0, stream>>>(hfin, ohe, out, n);
}
